// CustomMultiheadAttention_87780541596413
// MI455X (gfx1250) — matrix operand order repaired and checked by reading, not compiled, not run on hardware
//
#include <hip/hip_runtime.h>
#include <hip/hip_bf16.h>
#include <math.h>

// Shapes from the reference
#define BB 4
#define TT 1024
#define DD 1024
#define HH 16
#define HD 64

typedef __attribute__((ext_vector_type(16))) __bf16 bf16x16;
typedef __attribute__((ext_vector_type(4)))  __bf16 bf16x4;
typedef __attribute__((ext_vector_type(8)))  float  f32x8;

// ---- WMMA fragment loaders (CDNA5 wave32 layouts, CDNA5 ISA §7.12.2) ----

// A: 16x32 bf16 tile, element (m,k) at p[m*ld + k].
// lane L: m = L&15; elems 0..7 = K = 8*(L>>4)+0..7 ; elems 8..15 = K = 16+8*(L>>4)+0..7
__device__ __forceinline__ bf16x16 load_a_bf16(const __bf16* p, int ld, int lane) {
  const int m  = lane & 15;
  const int kb = (lane >> 4) << 3;
  const __bf16* row = p + (size_t)m * ld;
  bf16x16 a;
#pragma unroll
  for (int i = 0; i < 8; ++i) a[i] = row[kb + i];
#pragma unroll
  for (int i = 0; i < 8; ++i) a[8 + i] = row[16 + kb + i];
  return a;
}

// Same A layout, fp32 source, hardware cvt (fptrunc -> packed bf16 convert)
__device__ __forceinline__ bf16x16 load_a_f32(const float* p, int ld, int lane) {
  const int m  = lane & 15;
  const int kb = (lane >> 4) << 3;
  const float* row = p + (size_t)m * ld;
  bf16x16 a;
#pragma unroll
  for (int i = 0; i < 8; ++i) a[i] = (__bf16)row[kb + i];
#pragma unroll
  for (int i = 0; i < 8; ++i) a[8 + i] = (__bf16)row[16 + kb + i];
  return a;
}

// B: 32x16 (K x N) tile where element (k,n) lives at p[n*ld + k]  ("N-major" source:
// W stored [n][k] row-major, K rows stored [n][k], or V stored transposed [d][t]).
// lane L: n = L&15; elems 0..7 = K = 8*(L>>4)+0..7 ; elems 8..15 = K = 16+8*(L>>4)+0..7
__device__ __forceinline__ bf16x16 load_bT_bf16(const __bf16* p, int ld, int lane) {
  const int n  = lane & 15;
  const int kb = (lane >> 4) << 3;
  const __bf16* row = p + (size_t)n * ld + kb;
  bf16x16 bm;
#pragma unroll
  for (int i = 0; i < 8; ++i) bm[i] = row[i];
#pragma unroll
  for (int i = 0; i < 8; ++i) bm[8 + i] = row[16 + i];
  return bm;
}

#define WMMA_BF16(a, b, c) \
  __builtin_amdgcn_wmma_f32_16x16x32_bf16(false, (a), false, (b), (short)0, (c), false, false)

// ---- Kernel 0: bulk fp32 -> bf16 conversion (memory bound, one float4 per thread) ----
__global__ void __launch_bounds__(256)
cvt_f32_bf16_kernel(const float* __restrict__ src, __bf16* __restrict__ dst, int n4) {
  const int i = blockIdx.x * 256 + threadIdx.x;
  if (i < n4) {
    const float4 f = ((const float4*)src)[i];
    bf16x4 o;
    o[0] = (__bf16)f.x; o[1] = (__bf16)f.y; o[2] = (__bf16)f.z; o[3] = (__bf16)f.w;
    ((bf16x4*)dst)[i] = o;
  }
}

// ---- Kernel 1: input projection + head scatter.
// TRANS_OUT=false: Out[b,h,t,d]   (used for Q, K)
// TRANS_OUT=true : Out[b,h,d,t]   (used for V; makes the P.V B-operand contiguous)
template <bool TRANS_OUT>
__global__ void __launch_bounds__(256)
proj_head_kernel(const __bf16* __restrict__ Xb, const __bf16* __restrict__ Wb,
                 const float* __restrict__ bias, __bf16* __restrict__ Out) {
  const int lane = threadIdx.x & 31;
  const int wave = threadIdx.x >> 5;
  const int m0 = (blockIdx.x * 8 + wave) * 16;   // global row in [0, B*T)
  const int h  = blockIdx.y;                     // 64-wide N strip == one head
  const int n0 = h * HD;

  f32x8 acc[4] = { {}, {}, {}, {} };
  for (int kk = 0; kk < DD; kk += 32) {
    __builtin_prefetch(Xb + (size_t)m0 * DD + kk + 32, 0, 1);   // next A tile
    bf16x16 a = load_a_bf16(Xb + (size_t)m0 * DD + kk, DD, lane);
#pragma unroll
    for (int j = 0; j < 4; ++j) {
      bf16x16 bm = load_bT_bf16(Wb + (size_t)(n0 + j * 16) * DD + kk, DD, lane);
      acc[j] = WMMA_BF16(a, bm, acc[j]);
    }
  }
  const int nlo = lane & 15;
  const int mhi = (lane >> 4) << 3;
#pragma unroll
  for (int j = 0; j < 4; ++j) {
    const int n = j * 16 + nlo;                  // within-head dim
    const float bval = bias[n0 + n];
#pragma unroll
    for (int v = 0; v < 8; ++v) {
      const int row = m0 + mhi + v;
      const int b = row >> 10;                   // / TT
      const int t = row & (TT - 1);
      const __bf16 ov = (__bf16)(acc[j][v] + bval);
      if (TRANS_OUT)
        Out[(((size_t)(b * HH + h) * HD) + n) * TT + t] = ov;
      else
        Out[(((size_t)(b * HH + h) * TT) + t) * HD + n] = ov;
    }
  }
}

// ---- Kernel 2: raw scores S[bh,q,k] = (Q[bh,q,:] . K[bh,k,:]) / 8 ----
__global__ void __launch_bounds__(256)
scores_kernel(const __bf16* __restrict__ Qb, const __bf16* __restrict__ Kb,
              float* __restrict__ S) {
  const int lane = threadIdx.x & 31;
  const int wave = threadIdx.x >> 5;
  const int bh = blockIdx.z;
  const int q0 = (blockIdx.x * 8 + wave) * 16;
  const int k0 = blockIdx.y * 64;
  const __bf16* Qp = Qb + ((size_t)bh * TT + q0) * HD;

  f32x8 acc[4] = { {}, {}, {}, {} };
#pragma unroll
  for (int kk = 0; kk < HD; kk += 32) {
    bf16x16 a = load_a_bf16(Qp + kk, HD, lane);
#pragma unroll
    for (int j = 0; j < 4; ++j) {
      bf16x16 bm = load_bT_bf16(Kb + ((size_t)bh * TT + k0 + j * 16) * HD + kk, HD, lane);
      acc[j] = WMMA_BF16(a, bm, acc[j]);
    }
  }
  const float scale = 0.125f;                    // 1/sqrt(64)
  const int nlo = lane & 15;
  const int mhi = (lane >> 4) << 3;
#pragma unroll
  for (int j = 0; j < 4; ++j)
#pragma unroll
    for (int v = 0; v < 8; ++v) {
      const int q = q0 + mhi + v;
      const int k = k0 + j * 16 + nlo;
      S[((size_t)bh * TT + q) * TT + k] = acc[j][v] * scale;
    }
}

// ---- Kernel 3: masked softmax in place, one block per (b,h,q) row of 1024 ----
__global__ void __launch_bounds__(256)
softmax_kernel(float* __restrict__ S, const int* __restrict__ mask) {
  const int row = blockIdx.x;                    // bh*T + q
  const int b = row >> 14;                       // / (H*T) = / 16384
  float* rp = S + (size_t)row * TT;
  const int* mp = mask + b * TT;
  __shared__ float red[256];
  const int tid = threadIdx.x;

  float s[4]; int mk[4]; int any = 0;
#pragma unroll
  for (int i = 0; i < 4; ++i) {
    const int k = i * 256 + tid;
    s[i]  = rp[k];
    mk[i] = mp[k];
    any  |= (mk[i] == 0);
  }
  red[tid] = (float)any;
  __syncthreads();
  for (int off = 128; off > 0; off >>= 1) {
    if (tid < off) red[tid] = red[tid] + red[tid + off];
    __syncthreads();
  }
  const bool allMasked = (red[0] == 0.0f);
  __syncthreads();

  float v[4]; float mx = -3.4e38f;
#pragma unroll
  for (int i = 0; i < 4; ++i) {
    v[i] = allMasked ? -1.0e9f : (mk[i] ? -__builtin_inff() : s[i]);
    mx = fmaxf(mx, v[i]);
  }
  red[tid] = mx;
  __syncthreads();
  for (int off = 128; off > 0; off >>= 1) {
    if (tid < off) red[tid] = fmaxf(red[tid], red[tid + off]);
    __syncthreads();
  }
  mx = red[0];
  __syncthreads();

  float e[4]; float sum = 0.0f;
#pragma unroll
  for (int i = 0; i < 4; ++i) { e[i] = __expf(v[i] - mx); sum += e[i]; }
  red[tid] = sum;
  __syncthreads();
  for (int off = 128; off > 0; off >>= 1) {
    if (tid < off) red[tid] += red[tid + off];
    __syncthreads();
  }
  const float inv = 1.0f / red[0];
#pragma unroll
  for (int i = 0; i < 4; ++i) rp[i * 256 + tid] = e[i] * inv;
}

// ---- Kernel 4: AO[b,t,h*64+d] = sum_k P[bh,t,k] * Vt[bh,d,k]  (V pre-transposed) ----
__global__ void __launch_bounds__(256)
pv_kernel(const float* __restrict__ S, const __bf16* __restrict__ Vt,
          __bf16* __restrict__ AOb) {
  const int lane = threadIdx.x & 31;
  const int wave = threadIdx.x >> 5;
  const int bh = blockIdx.z;
  const int b = bh >> 4;
  const int h = bh & 15;
  const int q0 = (blockIdx.x * 8 + wave) * 16;
  const float* Pp = S + ((size_t)bh * TT + q0) * TT;
  const __bf16* Vp = Vt + (size_t)bh * HD * TT;   // [d][t] layout

  f32x8 acc[4] = { {}, {}, {}, {} };
  for (int kk = 0; kk < TT; kk += 32) {
    __builtin_prefetch(Pp + kk + 32, 0, 1);       // next A tile (fp32 stream)
    bf16x16 a = load_a_f32(Pp + kk, TT, lane);    // hw cvt, packed converts
#pragma unroll
    for (int j = 0; j < 4; ++j) {
      // element (k,n) at Vp[(j*16+n)*TT + kk + k] -> contiguous per lane
      bf16x16 bm = load_bT_bf16(Vp + (size_t)(j * 16) * TT + kk, TT, lane);
      acc[j] = WMMA_BF16(a, bm, acc[j]);
    }
  }
  const int nlo = lane & 15;
  const int mhi = (lane >> 4) << 3;
#pragma unroll
  for (int j = 0; j < 4; ++j)
#pragma unroll
    for (int v = 0; v < 8; ++v) {
      const int t  = q0 + mhi + v;
      const int dd = j * 16 + nlo;
      AOb[((size_t)(b * TT + t)) * DD + h * HD + dd] = (__bf16)acc[j][v];
    }
}

// ---- Kernel 5: output projection. Out[row,n] = AO[row,:] . Wo[n,:] + bo[n] (fp32) ----
__global__ void __launch_bounds__(256)
outproj_kernel(const __bf16* __restrict__ AOb, const __bf16* __restrict__ Wob,
               const float* __restrict__ bo, float* __restrict__ Out) {
  const int lane = threadIdx.x & 31;
  const int wave = threadIdx.x >> 5;
  const int m0 = (blockIdx.x * 8 + wave) * 16;
  const int n0 = blockIdx.y * 64;

  f32x8 acc[4] = { {}, {}, {}, {} };
  for (int kk = 0; kk < DD; kk += 32) {
    __builtin_prefetch(AOb + (size_t)m0 * DD + kk + 32, 0, 1);
    bf16x16 a = load_a_bf16(AOb + (size_t)m0 * DD + kk, DD, lane);
#pragma unroll
    for (int j = 0; j < 4; ++j) {
      bf16x16 bm = load_bT_bf16(Wob + (size_t)(n0 + j * 16) * DD + kk, DD, lane);
      acc[j] = WMMA_BF16(a, bm, acc[j]);
    }
  }
  const int nlo = lane & 15;
  const int mhi = (lane >> 4) << 3;
#pragma unroll
  for (int j = 0; j < 4; ++j) {
    const int n = n0 + j * 16 + nlo;
    const float bval = bo[n];
#pragma unroll
    for (int v = 0; v < 8; ++v) {
      const int row = m0 + mhi + v;
      Out[(size_t)row * DD + n] = acc[j][v] + bval;
    }
  }
}

extern "C" void kernel_launch(void* const* d_in, const int* in_sizes, int n_in,
                              void* d_out, int out_size, void* d_ws, size_t ws_size,
                              hipStream_t stream) {
  const float* query = (const float*)d_in[0];
  const float* key   = (const float*)d_in[1];
  const float* value = (const float*)d_in[2];
  const int*   mask  = (const int*)d_in[3];
  const float* Wq = (const float*)d_in[4];
  const float* bq = (const float*)d_in[5];
  const float* Wk = (const float*)d_in[6];
  const float* bk = (const float*)d_in[7];
  const float* Wv = (const float*)d_in[8];
  const float* bvv = (const float*)d_in[9];
  const float* Wo = (const float*)d_in[10];
  const float* bo = (const float*)d_in[11];

  float* out   = (float*)d_out;                          // [B,T,D]
  float* attnW = out + (size_t)BB * TT * DD;             // [B,H,Tq,Tk]

  // Workspace layout (bf16), 32M elements = 64 MB total:
  //   Xq|Xk|Xv [B*T,D], Wqb|Wkb|Wvb|Wob [D,D], Qb|Kb [B,H,T,64], Vt [B,H,64,T], AOb [B*T,D]
  const size_t NX = (size_t)BB * TT * DD;                // 4M
  const size_t NW = (size_t)DD * DD;                     // 1M
  __bf16* Xq  = (__bf16*)d_ws;
  __bf16* Xk  = Xq  + NX;
  __bf16* Xv  = Xk  + NX;
  __bf16* Wqb = Xv  + NX;
  __bf16* Wkb = Wqb + NW;
  __bf16* Wvb = Wkb + NW;
  __bf16* Wob = Wvb + NW;
  __bf16* Qb  = Wob + NW;
  __bf16* Kb  = Qb  + NX;
  __bf16* Vt  = Kb  + NX;
  __bf16* AOb = Vt  + NX;

  dim3 blk(256);

  // Stage all fp32 operands as bf16 once (memory bound, trivial cost)
  const int nx4 = (int)(NX / 4), nw4 = (int)(NW / 4);
  cvt_f32_bf16_kernel<<<dim3(nx4 / 256), blk, 0, stream>>>(query, Xq, nx4);
  cvt_f32_bf16_kernel<<<dim3(nx4 / 256), blk, 0, stream>>>(key,   Xk, nx4);
  cvt_f32_bf16_kernel<<<dim3(nx4 / 256), blk, 0, stream>>>(value, Xv, nx4);
  cvt_f32_bf16_kernel<<<dim3(nw4 / 256), blk, 0, stream>>>(Wq, Wqb, nw4);
  cvt_f32_bf16_kernel<<<dim3(nw4 / 256), blk, 0, stream>>>(Wk, Wkb, nw4);
  cvt_f32_bf16_kernel<<<dim3(nw4 / 256), blk, 0, stream>>>(Wv, Wvb, nw4);
  cvt_f32_bf16_kernel<<<dim3(nw4 / 256), blk, 0, stream>>>(Wo, Wob, nw4);

  dim3 gproj(BB * TT / 128, DD / 64);                    // (32, 16)
  proj_head_kernel<false><<<gproj, blk, 0, stream>>>(Xq, Wqb, bq,  Qb);
  proj_head_kernel<false><<<gproj, blk, 0, stream>>>(Xk, Wkb, bk,  Kb);
  proj_head_kernel<true ><<<gproj, blk, 0, stream>>>(Xv, Wvb, bvv, Vt);

  dim3 gsc(TT / 128, TT / 64, BB * HH);                  // (8, 16, 64)
  scores_kernel<<<gsc, blk, 0, stream>>>(Qb, Kb, attnW);

  softmax_kernel<<<dim3(BB * HH * TT), blk, 0, stream>>>(attnW, mask);

  pv_kernel<<<dim3(TT / 128, 1, BB * HH), blk, 0, stream>>>(attnW, Vt, AOb);

  outproj_kernel<<<gproj, blk, 0, stream>>>(AOb, Wob, bo, out);
}
